// CausalMultiheadSelfAttention_16131897164449
// MI455X (gfx1250) — matrix operand order repaired and checked by reading, not compiled, not run on hardware
//
#include <hip/hip_runtime.h>
#include <hip/hip_bf16.h>

typedef __attribute__((ext_vector_type(16))) __bf16 v16bf;
typedef __attribute__((ext_vector_type(8)))  __bf16 v8bf;
typedef __attribute__((ext_vector_type(8)))  float  v8f;

#define NUM_HEADS 12
#define D_MODEL   768
#define D_K       64
#define SEQ       2048
#define BATCH     4

// ---------------------------------------------------------------------------
// f32 -> bf16 elementwise conversion
// ---------------------------------------------------------------------------
__global__ void to_bf16_kernel(const float* __restrict__ in, __bf16* __restrict__ out, int n) {
    int i = blockIdx.x * blockDim.x + threadIdx.x;
    if (i < n) out[i] = (__bf16)in[i];
}

// ---------------------------------------------------------------------------
// GEMM: C[m][n] = sum_k A[m][k] * Bw[n][k]   (i.e. x @ W^T, W in [out][in])
// A: [M][K] bf16 row-major; Bw: [N=768][K] bf16 row-major.
// Tile 128x128, 256 threads = 8 waves (2x4), each wave owns 64x32 (4x2 WMMA tiles).
// mode 0: write bf16 to [B, H, S, D_K] layout (QKV projections)
// mode 1: write f32 row-major [M][768] (output projection)
// ---------------------------------------------------------------------------
__global__ __launch_bounds__(256) void gemm_xwT(const __bf16* __restrict__ A,
                                                const __bf16* __restrict__ Bw,
                                                void* __restrict__ Cout,
                                                int M, int K, int mode) {
    __shared__ __bf16 sA[128][40];   // 32 used + pad (row start 80B => 16B aligned)
    __shared__ __bf16 sB[128][40];

    const int tid  = threadIdx.x;
    const int lane = tid & 31;
    const int wave = tid >> 5;
    const int wr   = wave >> 2;              // 0..1 : row group of 64
    const int wc   = wave & 3;               // 0..3 : col group of 32
    const int l15  = lane & 15;
    const int hi   = (lane >= 16) ? 1 : 0;
    const int m0   = blockIdx.y * 128;
    const int n0   = blockIdx.x * 128;

    v8f acc[4][2];
#pragma unroll
    for (int i = 0; i < 4; ++i)
#pragma unroll
        for (int j = 0; j < 2; ++j) acc[i][j] = (v8f){0.f,0.f,0.f,0.f,0.f,0.f,0.f,0.f};

    const int srow = tid >> 1;               // 0..127
    const int shh  = tid & 1;                // k-half: 16 bf16 each

    for (int k0 = 0; k0 < K; k0 += 32) {
        // ---- stage A and B tiles (each thread copies 32B of each) ----
        {
            const uint4* ga = (const uint4*)(A + (size_t)(m0 + srow) * K + k0 + shh * 16);
            uint4* la = (uint4*)&sA[srow][shh * 16];
            la[0] = ga[0]; la[1] = ga[1];
            const uint4* gb = (const uint4*)(Bw + (size_t)(n0 + srow) * K + k0 + shh * 16);
            uint4* lb = (uint4*)&sB[srow][shh * 16];
            lb[0] = gb[0]; lb[1] = gb[1];
        }
        __syncthreads();

        // ---- load fragments ----
        v16bf aF[4], bF[2];
        const int ab = hi * 8;    // A swizzle: lanes>=16 take k offset +8 (and +24)
        const int kb = hi * 8;    // B swizzle: lanes>=16 take k offset +8 (and +24)
#pragma unroll
        for (int i = 0; i < 4; ++i) {
            const int mr = wr * 64 + i * 16 + l15;
            ((v8bf*)&aF[i])[0] = *(const v8bf*)&sA[mr][ab];
            ((v8bf*)&aF[i])[1] = *(const v8bf*)&sA[mr][16 + ab];
        }
#pragma unroll
        for (int j = 0; j < 2; ++j) {
            const int nr = wc * 32 + j * 16 + l15;
            ((v8bf*)&bF[j])[0] = *(const v8bf*)&sB[nr][kb];
            ((v8bf*)&bF[j])[1] = *(const v8bf*)&sB[nr][16 + kb];
        }
#pragma unroll
        for (int i = 0; i < 4; ++i)
#pragma unroll
            for (int j = 0; j < 2; ++j)
                acc[i][j] = __builtin_amdgcn_wmma_f32_16x16x32_bf16(
                    false, aF[i], false, bF[j], (short)0, acc[i][j], false, false);
        __syncthreads();
    }

    // ---- epilogue: C/D layout -> memory ----
#pragma unroll
    for (int i = 0; i < 4; ++i)
#pragma unroll
        for (int j = 0; j < 2; ++j)
#pragma unroll
            for (int r = 0; r < 8; ++r) {
                const int m = m0 + wr * 64 + i * 16 + r + hi * 8;
                const int n = n0 + wc * 32 + j * 16 + l15;
                const float v = acc[i][j][r];
                if (mode == 0) {
                    __bf16* o = (__bf16*)Cout;
                    const int b = m >> 11, s = m & 2047, h = n >> 6, d = n & 63;
                    o[((size_t)(b * NUM_HEADS + h) * SEQ + s) * D_K + d] = (__bf16)v;
                } else {
                    float* o = (float*)Cout;
                    o[(size_t)m * D_MODEL + n] = v;
                }
            }
}

// ---------------------------------------------------------------------------
// Flash attention: grid (SEQ/64, BATCH*NUM_HEADS), 128 threads = 4 waves.
// Each wave owns a 16-row query tile; workgroup covers 64 query rows.
// Loop over 32-key tiles with causal masking + online softmax.
// Q/K/V: bf16 [B,H,S,D_K]. Out: bf16 [B,S, H*D_K] (= [8192][768] row-major).
// ---------------------------------------------------------------------------
__global__ __launch_bounds__(128) void flash_attn(const __bf16* __restrict__ Q,
                                                  const __bf16* __restrict__ Km,
                                                  const __bf16* __restrict__ Vm,
                                                  __bf16* __restrict__ Out) {
    __shared__ __bf16 sK[32][72];        // [key][d]  (pitch 144B)
    __shared__ __bf16 sVt[64][40];       // [d][key]  transposed (pitch 80B)
    __shared__ __bf16 sP[4][16][40];     // per-wave probs [q][key] (pitch 80B)

    const int tid  = threadIdx.x;
    const int lane = tid & 31;
    const int wave = tid >> 5;
    const int l15  = lane & 15;
    const int hi   = (lane >= 16) ? 1 : 0;
    const int bh   = blockIdx.y;
    const int b    = bh / NUM_HEADS;
    const int h    = bh % NUM_HEADS;
    const size_t base = (size_t)bh * SEQ * D_K;
    const __bf16* Qh = Q + base;
    const __bf16* Kh = Km + base;
    const __bf16* Vh = Vm + base;
    const int q0 = blockIdx.x * 64 + wave * 16;

    // Q fragments for dk halves 0..31 and 32..63 (A-matrix swizzle)
    v16bf aq[2];
    {
        const __bf16* qp = Qh + (size_t)(q0 + l15) * D_K;
        const int ab = hi * 8;
        ((v8bf*)&aq[0])[0] = *(const v8bf*)(qp + 0  + ab);
        ((v8bf*)&aq[0])[1] = *(const v8bf*)(qp + 16 + ab);
        ((v8bf*)&aq[1])[0] = *(const v8bf*)(qp + 32 + ab);
        ((v8bf*)&aq[1])[1] = *(const v8bf*)(qp + 48 + ab);
    }

    float mstate[8], lstate[8];
    v8f accO[4];
#pragma unroll
    for (int r = 0; r < 8; ++r) { mstate[r] = -1e30f; lstate[r] = 0.f; }
#pragma unroll
    for (int d = 0; d < 4; ++d) accO[d] = (v8f){0.f,0.f,0.f,0.f,0.f,0.f,0.f,0.f};

    const int jend = blockIdx.x * 64 + 64;   // uniform per workgroup
    for (int j0 = 0; j0 < jend; j0 += 32) {
        // ---- stage K tile [32][64] ----
        {
            const int key  = tid >> 2;
            const int part = tid & 3;
            const uint4* gk = (const uint4*)(Kh + (size_t)(j0 + key) * D_K + part * 16);
            uint4* lk = (uint4*)&sK[key][part * 16];
            lk[0] = gk[0]; lk[1] = gk[1];
        }
        // ---- stage V transposed: sVt[d][key] ----
#pragma unroll
        for (int i = 0; i < 2; ++i) {
            const int idx = tid * 2 + i;       // 0..255
            const int key = idx >> 3;
            const int dc  = idx & 7;
            v8bf vv = *(const v8bf*)(Vh + (size_t)(j0 + key) * D_K + dc * 8);
#pragma unroll
            for (int jj = 0; jj < 8; ++jj) sVt[dc * 8 + jj][key] = vv[jj];
        }
        __syncthreads();

        // ---- scores: S[16 x 32] = Q(16x64) @ K^T, two 16x16 WMMA tiles ----
        v8f s[2];
        const int kb = hi * 8;
#pragma unroll
        for (int t = 0; t < 2; ++t) {
            const int nk = t * 16 + l15;
            v16bf bk0, bk1;
            ((v8bf*)&bk0)[0] = *(const v8bf*)&sK[nk][kb];
            ((v8bf*)&bk0)[1] = *(const v8bf*)&sK[nk][16 + kb];
            ((v8bf*)&bk1)[0] = *(const v8bf*)&sK[nk][32 + kb];
            ((v8bf*)&bk1)[1] = *(const v8bf*)&sK[nk][32 + 16 + kb];
            v8f z = (v8f){0.f,0.f,0.f,0.f,0.f,0.f,0.f,0.f};
            s[t] = __builtin_amdgcn_wmma_f32_16x16x32_bf16(false, aq[0], false, bk0, (short)0, z,    false, false);
            s[t] = __builtin_amdgcn_wmma_f32_16x16x32_bf16(false, aq[1], false, bk1, (short)0, s[t], false, false);
        }

        // ---- causal mask + online softmax (row = r + 8*hi, key col = lane&15) ----
        float scl[8], pr0[8], pr1[8];
#pragma unroll
        for (int r = 0; r < 8; ++r) {
            const int   qg  = q0 + r + hi * 8;
            const int   kg0 = j0 + l15;
            float s0 = s[0][r] * 0.125f;            // 1/sqrt(64)
            float s1 = s[1][r] * 0.125f;
            if (kg0      > qg) s0 = -1e30f;
            if (kg0 + 16 > qg) s1 = -1e30f;
            float rv = fmaxf(s0, s1);
            rv = fmaxf(rv, __shfl_xor(rv, 1, 32));
            rv = fmaxf(rv, __shfl_xor(rv, 2, 32));
            rv = fmaxf(rv, __shfl_xor(rv, 4, 32));
            rv = fmaxf(rv, __shfl_xor(rv, 8, 32));  // stays within 16-lane half
            const float mnew = fmaxf(mstate[r], rv);
            const float sc = __expf(mstate[r] - mnew);
            const float p0 = __expf(s0 - mnew);
            const float p1 = __expf(s1 - mnew);
            float rs = p0 + p1;
            rs += __shfl_xor(rs, 1, 32);
            rs += __shfl_xor(rs, 2, 32);
            rs += __shfl_xor(rs, 4, 32);
            rs += __shfl_xor(rs, 8, 32);
            lstate[r] = lstate[r] * sc + rs;
            mstate[r] = mnew;
            scl[r] = sc; pr0[r] = p0; pr1[r] = p1;
        }
#pragma unroll
        for (int d = 0; d < 4; ++d)
#pragma unroll
            for (int r = 0; r < 8; ++r) accO[d][r] *= scl[r];

        // ---- relayout P through LDS: C-layout regs -> A-fragment swizzle ----
#pragma unroll
        for (int r = 0; r < 8; ++r) {
            const int mrow = r + hi * 8;
            sP[wave][mrow][l15]      = (__bf16)pr0[r];
            sP[wave][mrow][16 + l15] = (__bf16)pr1[r];
        }
        __syncthreads();

        v16bf pF;
        {
            const int ab = hi * 8;
            ((v8bf*)&pF)[0] = *(const v8bf*)&sP[wave][l15][ab];
            ((v8bf*)&pF)[1] = *(const v8bf*)&sP[wave][l15][16 + ab];
        }
        // ---- O += P(16x32) @ V(32x64): four 16x16 output tiles ----
#pragma unroll
        for (int d = 0; d < 4; ++d) {
            const int dr = d * 16 + l15;
            v16bf vF;
            ((v8bf*)&vF)[0] = *(const v8bf*)&sVt[dr][kb];
            ((v8bf*)&vF)[1] = *(const v8bf*)&sVt[dr][16 + kb];
            accO[d] = __builtin_amdgcn_wmma_f32_16x16x32_bf16(
                false, pF, false, vF, (short)0, accO[d], false, false);
        }
        __syncthreads();
    }

    // ---- epilogue: normalize and write [B,S,H*D_K] bf16 ----
#pragma unroll
    for (int d = 0; d < 4; ++d)
#pragma unroll
        for (int r = 0; r < 8; ++r) {
            const int srow = q0 + r + hi * 8;
            const int dg   = d * 16 + l15;
            const float v  = accO[d][r] / lstate[r];
            Out[((size_t)(b * SEQ + srow)) * D_MODEL + h * D_K + dg] = (__bf16)v;
        }
}

// ---------------------------------------------------------------------------
// Host orchestration
// ---------------------------------------------------------------------------
extern "C" void kernel_launch(void* const* d_in, const int* in_sizes, int n_in,
                              void* d_out, int out_size, void* d_ws, size_t ws_size,
                              hipStream_t stream) {
    (void)in_sizes; (void)n_in; (void)out_size; (void)ws_size;
    const float* x  = (const float*)d_in[0];
    const float* wq = (const float*)d_in[1];
    const float* wk = (const float*)d_in[2];
    const float* wv = (const float*)d_in[3];
    const float* wo = (const float*)d_in[4];

    const int M  = BATCH * SEQ;            // 8192
    const int NX = M * D_MODEL;            // x / attn elements
    const int NW = D_MODEL * D_MODEL;      // weight elements
    const size_t QKV_BYTES = (size_t)BATCH * NUM_HEADS * SEQ * D_K * 2;

    char* ws = (char*)d_ws;
    size_t off = 0;
    auto take = [&](size_t bytes) { void* p = ws + off; off = (off + bytes + 255) & ~(size_t)255; return p; };

    __bf16* xb  = (__bf16*)take((size_t)NX * 2);
    __bf16* wqb = (__bf16*)take((size_t)NW * 2);
    __bf16* wkb = (__bf16*)take((size_t)NW * 2);
    __bf16* wvb = (__bf16*)take((size_t)NW * 2);
    __bf16* wob = (__bf16*)take((size_t)NW * 2);
    __bf16* Qb  = (__bf16*)take(QKV_BYTES);
    __bf16* Kb  = (__bf16*)take(QKV_BYTES);
    __bf16* Vb  = (__bf16*)take(QKV_BYTES);
    __bf16* attnb = xb;   // alias: xb is dead after the V projection

    // 1) convert inputs to bf16
    to_bf16_kernel<<<(NX + 255) / 256, 256, 0, stream>>>(x,  xb,  NX);
    to_bf16_kernel<<<(NW + 255) / 256, 256, 0, stream>>>(wq, wqb, NW);
    to_bf16_kernel<<<(NW + 255) / 256, 256, 0, stream>>>(wk, wkb, NW);
    to_bf16_kernel<<<(NW + 255) / 256, 256, 0, stream>>>(wv, wvb, NW);
    to_bf16_kernel<<<(NW + 255) / 256, 256, 0, stream>>>(wo, wob, NW);

    // 2) QKV projections (write into [B,H,S,D_K])
    dim3 ggrid(D_MODEL / 128, M / 128);    // (6, 64)
    gemm_xwT<<<ggrid, 256, 0, stream>>>(xb, wqb, Qb, M, D_MODEL, 0);
    gemm_xwT<<<ggrid, 256, 0, stream>>>(xb, wkb, Kb, M, D_MODEL, 0);
    gemm_xwT<<<ggrid, 256, 0, stream>>>(xb, wvb, Vb, M, D_MODEL, 0);

    // 3) causal flash attention -> attn [B,S,H*D_K] bf16
    dim3 agrid(SEQ / 64, BATCH * NUM_HEADS);   // (32, 48)
    flash_attn<<<agrid, 128, 0, stream>>>(Qb, Kb, Vb, attnb);

    // 4) output projection -> f32 d_out
    gemm_xwT<<<ggrid, 256, 0, stream>>>(attnb, wob, d_out, M, D_MODEL, 1);
}
